// SPINN_42863773614899
// MI455X (gfx1250) — matrix operand order repaired and checked by reading, not compiled, not run on hardware
//
#include <hip/hip_runtime.h>
#include <hip/hip_bf16.h>

// ---------------------------------------------------------------------------
// SPINN on MI455X (gfx1250): persistent wave32 WMMA kernel.
//   L=64 steps, B=256 batch, H=256, DT=64, STACK=6
//   tracker GEMM : x(16x832)  @ Wg^T(832x256)   bf16 -> f32 WMMA 16x16x32
//   treeLSTM GEMM: x(16x576)  @ Wl^T(576x1280)  bf16 -> f32 WMMA 16x16x32
// Each workgroup owns 16 batch rows end-to-end (batch rows are independent).
// ---------------------------------------------------------------------------

typedef __attribute__((ext_vector_type(16))) __bf16 v16bf;
typedef __attribute__((ext_vector_type(8)))  __bf16 v8bf;
typedef __attribute__((ext_vector_type(8)))  float  v8f;

#define SHIFT_T 3
#define REDUCE_T 2

constexpr int Lsteps = 64;
constexpr int Bsz    = 256;
constexpr int H      = 256;
constexpr int DT     = 64;
constexpr int STACKN = 6;

constexpr int KG  = 3 * H + DT;      // 832  (buf_h | s1h | s2h | th_prev)
constexpr int KL  = 2 * H + DT;      // 576  (lh=s2h | rh=s1h | th_new)
constexpr int KGT = KG / 32;         // 26 k-tiles
constexpr int KLT = KL / 32;         // 18 k-tiles
constexpr int NGT = (4 * DT) / 16;   // 16 n-tiles (tracker, 4 gates x 4 tiles)
constexpr int NLT = (5 * H) / 16;    // 80 n-tiles (lstm, 5 gates x 16 tiles)
constexpr int MT  = 16;              // batch rows per workgroup
constexpr int WGS = 512;             // 16 waves

constexpr size_t STACK_ELEMS = (size_t)Bsz * STACKN * (2 * H);   // f32
constexpr size_t WG_ELEMS    = (size_t)(4 * DT) * KG;            // bf16
constexpr size_t WL_ELEMS    = (size_t)(5 * H) * KL;             // bf16

__device__ __forceinline__ float sigf(float x) {
    return 1.0f / (1.0f + __expf(-x));   // v_exp_f32 path, branch-free
}

// gfx1250 has a V_TANH_F32 transcendental; use it if the builtin exists,
// otherwise a branch-free exp-based formulation (never the branchy libm tanhf,
// which forces saveexec divergence + register spills in the hot tail).
#if defined(__has_builtin)
#if __has_builtin(__builtin_amdgcn_tanhf)
#define FAST_TANH(x) __builtin_amdgcn_tanhf(x)
#endif
#endif
#ifndef FAST_TANH
#define FAST_TANH(x) (2.0f * sigf(2.0f * (x)) - 1.0f)
#endif

// A-fragment (16x32 bf16) from row-major bf16 panel in LDS.
// ISA layout: lanes 0-15 -> M=lane, elems 0-7 = K[8*half .. +7],
//             elems 8-15 = K[16+8*half .. +7]  (half = lane>>4)
__device__ __forceinline__ v16bf load_a_frag(const __bf16* xs, int strideElems,
                                             int kt, int lane) {
    const int m    = lane & 15;
    const int half = lane >> 4;
    const __bf16* p = xs + (size_t)m * strideElems + kt * 32 + half * 8;
    v8bf lo = *reinterpret_cast<const v8bf*>(p);
    v8bf hi = *reinterpret_cast<const v8bf*>(p + 16);
    return __builtin_shufflevector(lo, hi, 0, 1, 2, 3, 4, 5, 6, 7,
                                   8, 9, 10, 11, 12, 13, 14, 15);
}

// B-fragment (32x16 bf16) pre-packed: addr = ((nt*KT + kt)*32 + lane)*16
__device__ __forceinline__ v16bf load_b_frag(const __bf16* W, int numKT,
                                             int nt, int kt, int lane) {
    const __bf16* p = W + ((size_t)(nt * numKT + kt) * 32 + lane) * 16;
    return *reinterpret_cast<const v16bf*>(p);
}

#define WMMA_BF16(A, B, C) \
    __builtin_amdgcn_wmma_f32_16x16x32_bf16(false, (A), false, (B), (short)0, (C), false, false)

// ---------------------------------------------------------------------------
// Kernel 1: pack weights f32 -> bf16 B-fragment order, fuse K dims, sum biases
// ---------------------------------------------------------------------------
__global__ void spinn_pack(const float* __restrict__ W_ih,
                           const float* __restrict__ W_hh,
                           const float* __restrict__ b_ih,
                           const float* __restrict__ b_hh,
                           const float* __restrict__ W_left,
                           const float* __restrict__ W_right,
                           const float* __restrict__ W_track,
                           __bf16* __restrict__ Wg,
                           __bf16* __restrict__ Wl,
                           float* __restrict__ bg) {
    const int NGE = NGT * KGT * 512;   // 212992
    const int NLE = NLT * KLT * 512;   // 737280
    int t = blockIdx.x * blockDim.x + threadIdx.x;
    if (t < NGE) {
        int e = t & 15, lane = (t >> 4) & 31, frag = t >> 9;
        int kt = frag % KGT, nt = frag / KGT;
        int n = nt * 16 + (lane & 15);
        int k = kt * 32 + (lane >> 4) * 8 + (e & 7) + ((e >> 3) << 4);
        float v = (k < 3 * H) ? W_ih[(size_t)n * (3 * H) + k]
                              : W_hh[(size_t)n * DT + (k - 3 * H)];
        Wg[t] = (__bf16)v;
    } else if (t < NGE + NLE) {
        int u = t - NGE;
        int e = u & 15, lane = (u >> 4) & 31, frag = u >> 9;
        int kt = frag % KLT, nt = frag / KLT;
        int n = nt * 16 + (lane & 15);
        int k = kt * 32 + (lane >> 4) * 8 + (e & 7) + ((e >> 3) << 4);
        float v = (k < H)     ? W_left[(size_t)n * H + k]
                : (k < 2 * H) ? W_right[(size_t)n * H + (k - H)]
                              : W_track[(size_t)n * DT + (k - 2 * H)];
        Wl[u] = (__bf16)v;
    } else if (t < NGE + NLE + 4 * DT) {
        int n = t - NGE - NLE;
        bg[n] = b_ih[n] + b_hh[n];
    }
}

// ---------------------------------------------------------------------------
// Kernel 2: persistent scan. One block = 16 batch rows, 16 waves.
// ---------------------------------------------------------------------------
__global__ void __launch_bounds__(WGS)
spinn_main(const float* __restrict__ buffers,     // (L, B, 2H) f32
           const int*   __restrict__ transitions, // (L, B) i32
           const float* __restrict__ b_left,      // (5H)
           const __bf16* __restrict__ Wg,
           const __bf16* __restrict__ Wl,
           const float* __restrict__ bg,          // (4*DT) fused bias
           float* __restrict__ stackG,            // (B, STACK, 2H) f32 scratch
           float* __restrict__ out) {             // (B,256) ++ (B,128)
    __shared__ __align__(32) __bf16 xg[MT * KG];  // 26.0 KB
    __shared__ __align__(32) __bf16 xl[MT * KL];  // 18.0 KB
    __shared__ float tcS[MT * DT];                //  4 KB
    __shared__ float thS[MT * DT];                //  4 KB
    __shared__ int   sptrS[MT];
    __shared__ int   bptrS[MT];

    const int tid  = threadIdx.x;
    const int wave = tid >> 5;
    const int lane = tid & 31;
    const int ln   = lane & 15;
    const int half = lane >> 4;
    const int gb   = blockIdx.x * MT;   // global batch base for this WG

    // Loop-invariant per-lane output column + biases (hoisted off the 64-step
    // serial critical path).
    const int nL = wave * 16 + ln;      // tree-LSTM column (0..255)
    const float Ba  = b_left[nL];
    const float Bi  = b_left[H + nL];
    const float Bf1 = b_left[2 * H + nL];
    const float Bf2 = b_left[3 * H + nL];
    const float Bo  = b_left[4 * H + nL];
    const int nT = (wave & 3) * 16 + ln; // tracker column (0..63), waves 0..3
    const float Bgi = bg[nT];
    const float Bgf = bg[DT + nT];
    const float Bgg = bg[2 * DT + nT];
    const float Bgo = bg[3 * DT + nT];

    // ---- init: stack slots 0,1 = buffers[0]; sptr=2; bptr=L; th=tc=0 ----
    for (int e = tid; e < MT * 512; e += WGS) {
        int m = e >> 9, col = e & 511;
        float v = buffers[(size_t)(gb + m) * 512 + col];
        size_t sb = (size_t)(gb + m) * STACKN * 512;
        stackG[sb + col]       = v;
        stackG[sb + 512 + col] = v;
    }
    for (int e = tid; e < MT * DT; e += WGS) {
        tcS[e] = 0.0f;
        thS[e] = 0.0f;
        int m = e >> 6, n = e & 63;
        xg[m * KG + 3 * H + n] = (__bf16)0.0f;   // th_prev section, step 0
    }
    if (tid < MT) { sptrS[tid] = 2; bptrS[tid] = Lsteps; }
    __threadfence_block();
    __syncthreads();

    for (int step = 0; step < Lsteps; ++step) {
        // ---- Phase 1: build bf16 activation panels -----------------------
        for (int e = tid; e < MT * H; e += WGS) {
            int m = e >> 8, col = e & 255;
            int sp = sptrS[m], bp = bptrS[m];
            float bh = buffers[((size_t)(bp - 1) * Bsz + gb + m) * 512 + col];
            size_t sb = (size_t)(gb + m) * STACKN * 512;
            float h1 = stackG[sb + (size_t)(sp - 1) * 512 + col];  // s1.h
            float h2 = stackG[sb + (size_t)(sp - 2) * 512 + col];  // s2.h
            xg[m * KG + col]           = (__bf16)bh;
            xg[m * KG + H + col]       = (__bf16)h1;
            xg[m * KG + 2 * H + col]   = (__bf16)h2;
            xl[m * KL + col]           = (__bf16)h2;   // lh = s2.h
            xl[m * KL + H + col]       = (__bf16)h1;   // rh = s1.h
        }
        __syncthreads();

        // ---- Phase 2: tracker GEMM + LSTM cell (waves 0..3) --------------
        if (wave < 4) {
            v8f ai = {}, af = {}, ag = {}, ao = {};
            for (int kt = 0; kt < KGT; ++kt) {
                v16bf a   = load_a_frag(xg, KG, kt, lane);
                v16bf bi  = load_b_frag(Wg, KGT, 0 * 4 + wave, kt, lane);
                v16bf bf  = load_b_frag(Wg, KGT, 1 * 4 + wave, kt, lane);
                v16bf bgv = load_b_frag(Wg, KGT, 2 * 4 + wave, kt, lane);
                v16bf bo  = load_b_frag(Wg, KGT, 3 * 4 + wave, kt, lane);
                ai = WMMA_BF16(a, bi, ai);
                af = WMMA_BF16(a, bf, af);
                ag = WMMA_BF16(a, bgv, ag);
                ao = WMMA_BF16(a, bo, ao);
            }
#pragma unroll
            for (int r = 0; r < 8; ++r) {
                int m = r + 8 * half;
                float gi = ai[r] + Bgi, gf = af[r] + Bgf;
                float gg = ag[r] + Bgg, go = ao[r] + Bgo;
                float tc = tcS[m * DT + nT];
                tc = sigf(gf) * tc + sigf(gi) * FAST_TANH(gg);
                float th = sigf(go) * FAST_TANH(tc);
                tcS[m * DT + nT] = tc;
                thS[m * DT + nT] = th;
                xl[m * KL + 2 * H + nT] = (__bf16)th;  // th for this step's LSTM
                xg[m * KG + 3 * H + nT] = (__bf16)th;  // th_prev for next step
            }
        }
        __syncthreads();

        // ---- Phase 3: tree-LSTM GEMM + in-register cell + reduce write ---
        {
            const int cg = wave;                      // column group 0..15
            v8f aa = {}, aI = {}, aF1 = {}, aF2 = {}, aO = {};
            for (int kt = 0; kt < KLT; ++kt) {
                v16bf a = load_a_frag(xl, KL, kt, lane);
                if (kt + 1 < KLT)   // stream next k-tile of gate-a weights
                    __builtin_prefetch(
                        Wl + ((size_t)(cg * KLT + kt + 1) * 32 + lane) * 16, 0, 1);
                v16bf b0 = load_b_frag(Wl, KLT, 0 * 16 + cg, kt, lane);
                v16bf b1 = load_b_frag(Wl, KLT, 1 * 16 + cg, kt, lane);
                v16bf b2 = load_b_frag(Wl, KLT, 2 * 16 + cg, kt, lane);
                v16bf b3 = load_b_frag(Wl, KLT, 3 * 16 + cg, kt, lane);
                v16bf b4 = load_b_frag(Wl, KLT, 4 * 16 + cg, kt, lane);
                aa  = WMMA_BF16(a, b0, aa);
                aI  = WMMA_BF16(a, b1, aI);
                aF1 = WMMA_BF16(a, b2, aF1);
                aF2 = WMMA_BF16(a, b3, aF2);
                aO  = WMMA_BF16(a, b4, aO);
            }
#pragma unroll
            for (int r = 0; r < 8; ++r) {
                int m = r + 8 * half;
                int tr = transitions[step * Bsz + gb + m];
                float av = FAST_TANH(aa[r] + Ba);
                float iv = sigf(aI[r] + Bi);
                float f1 = sigf(aF1[r] + Bf1);
                float f2 = sigf(aF2[r] + Bf2);
                float ov = sigf(aO[r] + Bo);
                int sp = sptrS[m];
                size_t sb = (size_t)(gb + m) * STACKN * 512;
                float lc = stackG[sb + (size_t)(sp - 2) * 512 + H + nL];
                float rc = stackG[sb + (size_t)(sp - 1) * 512 + H + nL];
                float c = av * iv + f1 * lc + f2 * rc;
                float h = ov * FAST_TANH(c);
                if (tr == REDUCE_T) {
                    stackG[sb + (size_t)(sp - 2) * 512 + nL]     = h;
                    stackG[sb + (size_t)(sp - 2) * 512 + H + nL] = c;
                }
            }
        }
        __syncthreads();

        // ---- Phase 4: shift pushes (copy buf_top onto stack) -------------
        for (int e = tid; e < MT * 512; e += WGS) {
            int m = e >> 9, col = e & 511;
            int tr = transitions[step * Bsz + gb + m];
            if (tr == SHIFT_T) {
                int sp = sptrS[m], bp = bptrS[m];
                stackG[(size_t)(gb + m) * STACKN * 512 + (size_t)sp * 512 + col] =
                    buffers[((size_t)(bp - 1) * Bsz + gb + m) * 512 + col];
            }
        }
        __syncthreads();

        // ---- Phase 5: pointer updates ------------------------------------
        if (tid < MT) {
            int tr = transitions[step * Bsz + gb + tid];
            if (tr == SHIFT_T)      { sptrS[tid] += 1; bptrS[tid] -= 1; }
            else if (tr == REDUCE_T) { sptrS[tid] -= 1; }
        }
        __threadfence_block();
        __syncthreads();
    }

    // ---- Final outputs: out = top-of-stack h ; tracker_state = [th|tc] ----
    for (int e = tid; e < MT * H; e += WGS) {
        int m = e >> 8, col = e & 255;
        int sp = sptrS[m];
        out[(size_t)(gb + m) * H + col] =
            stackG[(size_t)(gb + m) * STACKN * 512 + (size_t)(sp - 1) * 512 + col];
    }
    for (int e = tid; e < MT * 2 * DT; e += WGS) {
        int m = e >> 7, col = e & 127;
        float v = (col < DT) ? thS[m * DT + col] : tcS[m * DT + (col - DT)];
        out[(size_t)Bsz * H + (size_t)(gb + m) * (2 * DT) + col] = v;
    }
}

// ---------------------------------------------------------------------------
extern "C" void kernel_launch(void* const* d_in, const int* in_sizes, int n_in,
                              void* d_out, int out_size, void* d_ws, size_t ws_size,
                              hipStream_t stream) {
    const float* buffers = (const float*)d_in[0];
    const int*   trans   = (const int*)d_in[1];
    const float* W_left  = (const float*)d_in[2];
    const float* b_left  = (const float*)d_in[3];
    const float* W_right = (const float*)d_in[4];
    const float* W_track = (const float*)d_in[5];
    const float* W_ih    = (const float*)d_in[6];
    const float* b_ih    = (const float*)d_in[7];
    const float* W_hh    = (const float*)d_in[8];
    const float* b_hh    = (const float*)d_in[9];
    float* out = (float*)d_out;

    // workspace layout: [stack f32][Wg bf16][Wl bf16][bg f32]
    float*  stackG = (float*)d_ws;
    __bf16* Wg     = (__bf16*)((char*)d_ws + STACK_ELEMS * sizeof(float));
    __bf16* Wl     = Wg + WG_ELEMS;
    float*  bgp    = (float*)(Wl + WL_ELEMS);

    const int packTotal = NGT * KGT * 512 + NLT * KLT * 512 + 4 * DT;
    spinn_pack<<<(packTotal + 255) / 256, 256, 0, stream>>>(
        W_ih, W_hh, b_ih, b_hh, W_left, W_right, W_track, Wg, Wl, bgp);

    spinn_main<<<Bsz / MT, WGS, 0, stream>>>(
        buffers, trans, b_left, Wg, Wl, bgp, stackG, out);
}
